// Transformer_74440373174611
// MI455X (gfx1250) — matrix operand order repaired and checked by reading, not compiled, not run on hardware
//
#include <hip/hip_runtime.h>
#include <hip/hip_bf16.h>

// Point-transformer layer for MI455X (gfx1250, wave32).
// Kernel 1: QKV projections via v_wmma_f32_16x16x32_f16, LDS weights stored
//           column-major + padded so B fragments are contiguous, conflict-free
//           ds_load_b128 pairs (was: 16x conflicted ds_load_u16 per fragment).
// Kernel 2: L2-resident KNN gather + subtraction attention (VALU, memory-bound).

#define KNN    16
#define EPS    1e-5f

typedef __attribute__((ext_vector_type(16))) _Float16 v16h;
typedef __attribute__((ext_vector_type(8)))  _Float16 v8h;
typedef __attribute__((ext_vector_type(8)))  float    v8f;

#define WSTRIDE 72            // padded K-stride (f16 elems): 144B = 36 dwords,
                              // 16 lanes x b128 tile all 64 LDS banks exactly.
#define WMAT    (64 * WSTRIDE)

// ---------------------------------------------------------------------------
// Kernel 1: feat(N x 64) @ {Wq,Wk,Wv}(64 x 64) + bias -> fq, fk, fv (fp32)
// Block = 256 threads = 8 waves; each wave computes TWO 16-row tiles (32 rows)
// so each B fragment feeds 2 WMMAs. 48 WMMAs / wave.
// ---------------------------------------------------------------------------
__global__ __launch_bounds__(256) void qkv_wmma_kernel(
    const float* __restrict__ feat,
    const float* __restrict__ Wq, const float* __restrict__ bq,
    const float* __restrict__ Wk, const float* __restrict__ bk,
    const float* __restrict__ Wv, const float* __restrict__ bv,
    float* __restrict__ fq, float* __restrict__ fk, float* __restrict__ fv)
{
    // Column-major, padded: sW[mat*WMAT + col*WSTRIDE + k]  (27 KB)
    __shared__ _Float16 sW[3 * WMAT];

    const int tid = threadIdx.x;
    #pragma unroll
    for (int i = 0; i < 16; ++i) {
        const int e = tid + i * 256;            // 4096 elements per matrix
        const int k = e >> 6;
        const int c = e & 63;
        const int d = c * WSTRIDE + k;
        sW[d]            = (_Float16)Wq[e];
        sW[WMAT + d]     = (_Float16)Wk[e];
        sW[2 * WMAT + d] = (_Float16)Wv[e];
    }
    __syncthreads();

    const int wave = tid >> 5;
    const int lane = tid & 31;
    const bool hi  = lane >= 16;
    const int m    = lane & 15;
    const int row0 = blockIdx.x * 256 + wave * 32;   // 2 M-tiles per wave

    // A fragments (16x32 f16, ISA layout):
    //   lanes 0-15 : elems[0..7]=K+0..7,  elems[8..15]=K+16..23
    //   lanes 16-31: elems[0..7]=K+8..15, elems[8..15]=K+24..31
    v16h a[2][2];                                    // [tile][kstep]
    #pragma unroll
    for (int tile = 0; tile < 2; ++tile) {
        const int row = row0 + tile * 16 + m;
        #pragma unroll
        for (int step = 0; step < 2; ++step) {
            const int base = step * 32 + (hi ? 8 : 0);
            float tf[16];
            ((float4*)tf)[0] = *(const float4*)(feat + row * 64 + base);
            ((float4*)tf)[1] = *(const float4*)(feat + row * 64 + base + 4);
            ((float4*)tf)[2] = *(const float4*)(feat + row * 64 + base + 16);
            ((float4*)tf)[3] = *(const float4*)(feat + row * 64 + base + 20);
            #pragma unroll
            for (int i = 0; i < 16; ++i) a[tile][step][i] = (_Float16)tf[i];
        }
    }

    const float* const bias[3] = { bq, bk, bv };
    float* const       outp[3] = { fq, fk, fv };
    const int koff = hi ? 8 : 0;    // B frag: lanes 0-15 K=kk+0..7,16..23; 16-31 K=kk+8..15,24..31

    #pragma unroll
    for (int mat = 0; mat < 3; ++mat) {
        #pragma unroll
        for (int nt = 0; nt < 4; ++nt) {
            const int col = nt * 16 + m;
            const _Float16* wp = sW + mat * WMAT + col * WSTRIDE + koff;

            // Two contiguous 16-byte LDS loads per fragment (conflict-free).
            const v8h b0lo = *(const v8h*)(wp);
            const v8h b0hi = *(const v8h*)(wp + 16);
            const v8h b1lo = *(const v8h*)(wp + 32);
            const v8h b1hi = *(const v8h*)(wp + 48);
            const v16h b0 = __builtin_shufflevector(b0lo, b0hi,
                0,1,2,3,4,5,6,7,8,9,10,11,12,13,14,15);
            const v16h b1 = __builtin_shufflevector(b1lo, b1hi,
                0,1,2,3,4,5,6,7,8,9,10,11,12,13,14,15);

            const float bb  = bias[mat][col];
            float* const op = outp[mat];

            #pragma unroll
            for (int tile = 0; tile < 2; ++tile) {
                v8f acc = {};
                acc = __builtin_amdgcn_wmma_f32_16x16x32_f16(
                          false, a[tile][0], false, b0, (short)0, acc, false, false);
                acc = __builtin_amdgcn_wmma_f32_16x16x32_f16(
                          false, a[tile][1], false, b1, (short)0, acc, false, false);
                // C/D layout: VGPR r -> M=r (lanes 0-15) / M=r+8 (lanes 16-31)
                const int rbase = row0 + tile * 16 + (hi ? 8 : 0);
                #pragma unroll
                for (int r = 0; r < 8; ++r)
                    op[(size_t)(rbase + r) * 64 + col] = acc[r] + bb;
            }
        }
    }
}

// ---------------------------------------------------------------------------
// Kernel 2: gather + position MLP + subtraction attention + weighted sum.
// One wave32 per point; lane owns channels (2*lane, 2*lane+1) -> float2
// gathers are fully coalesced (256B per gathered row). Reduction 64->8 via
// shfl_xor butterfly; the 8x8 stage and softmax are lane-redundant.
// ---------------------------------------------------------------------------
__global__ __launch_bounds__(256) void attn_kernel(
    const float* __restrict__ point, const int* __restrict__ idx,
    const float* __restrict__ fq, const float* __restrict__ fk,
    const float* __restrict__ fv,
    const float* __restrict__ p_w1, const float* __restrict__ p_b1,
    const float* __restrict__ p_bn_g, const float* __restrict__ p_bn_b,
    const float* __restrict__ p_bn_m, const float* __restrict__ p_bn_v,
    const float* __restrict__ p_w2, const float* __restrict__ p_b2,
    const float* __restrict__ bn1_g, const float* __restrict__ bn1_b,
    const float* __restrict__ bn1_m, const float* __restrict__ bn1_v,
    const float* __restrict__ w_w1, const float* __restrict__ w_b1,
    const float* __restrict__ bn2_g, const float* __restrict__ bn2_b,
    const float* __restrict__ bn2_m, const float* __restrict__ bn2_v,
    const float* __restrict__ w_w2, const float* __restrict__ w_b2,
    float* __restrict__ out)
{
    const int tid  = threadIdx.x;
    const int n    = blockIdx.x * 8 + (tid >> 5);
    const int lane = tid & 31;
    const int c0   = lane * 2;
    const int c1   = c0 + 1;

    // ---- fused small parameters (registers; uniform across lanes) ----
    float w1f[9], b1f[3];
    #pragma unroll
    for (int s = 0; s < 3; ++s) {
        const float sc = p_bn_g[s] * rsqrtf(p_bn_v[s] + EPS);
        w1f[0 * 3 + s] = p_w1[0 * 3 + s] * sc;
        w1f[1 * 3 + s] = p_w1[1 * 3 + s] * sc;
        w1f[2 * 3 + s] = p_w1[2 * 3 + s] * sc;
        b1f[s] = p_b1[s] * sc + p_bn_b[s] - p_bn_m[s] * sc;
    }
    float wb1[8], s2[8], bb2[8], wb2[8];
    #pragma unroll
    for (int r = 0; r < 8; ++r) {
        wb1[r] = w_b1[r];
        const float sc = bn2_g[r] * rsqrtf(bn2_v[r] + EPS);
        s2[r]  = sc;
        bb2[r] = bn2_b[r] - bn2_m[r] * sc;
        wb2[r] = w_b2[r];
    }
    float rw2[64];
    #pragma unroll
    for (int i = 0; i < 64; ++i) rw2[i] = w_w2[i];

    // ---- per-lane channel-specific parameters ----
    const float sc10 = bn1_g[c0] * rsqrtf(bn1_v[c0] + EPS);
    const float sb10 = bn1_b[c0] - bn1_m[c0] * sc10;
    const float sc11 = bn1_g[c1] * rsqrtf(bn1_v[c1] + EPS);
    const float sb11 = bn1_b[c1] - bn1_m[c1] * sc11;

    float pw2a[3], pw2b[3];
    #pragma unroll
    for (int r = 0; r < 3; ++r) {
        pw2a[r] = p_w2[r * 64 + c0];
        pw2b[r] = p_w2[r * 64 + c1];
    }
    const float pb2a = p_b2[c0], pb2b = p_b2[c1];

    float ww1a[8], ww1b[8];
    #pragma unroll
    for (int r = 0; r < 8; ++r) {
        ww1a[r] = w_w1[c0 * 8 + r];
        ww1b[r] = w_w1[c1 * 8 + r];
    }

    const float px = point[n * 3 + 0];
    const float py = point[n * 3 + 1];
    const float pz = point[n * 3 + 2];
    const float2 q = ((const float2*)fq)[(size_t)n * 32 + lane];

    float acc0 = 0.f, acc1 = 0.f;

    for (int k = 0; k < KNN; ++k) {
        const int j = idx[n * KNN + k];
        if (k + 1 < KNN) {
            const int jn = idx[n * KNN + k + 1];
            __builtin_prefetch(fk + (size_t)jn * 64, 0, 0);   // global_prefetch_b8
            __builtin_prefetch(fv + (size_t)jn * 64, 0, 0);
        }

        // relative position + linear_p stage 1 (Dense(3)+BN+ReLU, fused)
        const float dx = point[j * 3 + 0] - px;
        const float dy = point[j * 3 + 1] - py;
        const float dz = point[j * 3 + 2] - pz;
        const float h0 = fmaxf(0.f, dx * w1f[0] + dy * w1f[3] + dz * w1f[6] + b1f[0]);
        const float h1 = fmaxf(0.f, dx * w1f[1] + dy * w1f[4] + dz * w1f[7] + b1f[1]);
        const float h2 = fmaxf(0.f, dx * w1f[2] + dy * w1f[5] + dz * w1f[8] + b1f[2]);
        // linear_p stage 2 for this lane's two channels
        const float p0 = pb2a + h0 * pw2a[0] + h1 * pw2a[1] + h2 * pw2a[2];
        const float p1 = pb2b + h0 * pw2b[0] + h1 * pw2b[1] + h2 * pw2b[2];

        // gather k/v rows (coalesced float2 per lane)
        const float2 kk = ((const float2*)fk)[(size_t)j * 32 + lane];
        const float2 vv = ((const float2*)fv)[(size_t)j * 32 + lane];

        // w = relu(bn1(g_k - q + p))
        const float wp0 = fmaxf(0.f, (kk.x - q.x + p0) * sc10 + sb10);
        const float wp1 = fmaxf(0.f, (kk.y - q.y + p1) * sc11 + sb11);

        // 64 -> 8 projection: per-lane partials, then wave32 butterfly sum
        float part[8];
        #pragma unroll
        for (int r = 0; r < 8; ++r)
            part[r] = wp0 * ww1a[r] + wp1 * ww1b[r];
        #pragma unroll
        for (int off = 16; off >= 1; off >>= 1) {
            #pragma unroll
            for (int r = 0; r < 8; ++r)
                part[r] += __shfl_xor(part[r], off, 32);
        }

        // BN + ReLU, then 8 -> 8 dense (lane-redundant; values are uniform)
        float hh[8];
        #pragma unroll
        for (int r = 0; r < 8; ++r)
            hh[r] = fmaxf(0.f, (part[r] + wb1[r]) * s2[r] + bb2[r]);

        float t[8];
        #pragma unroll
        for (int o = 0; o < 8; ++o) {
            float s = wb2[o];
            #pragma unroll
            for (int r = 0; r < 8; ++r) s += hh[r] * rw2[r * 8 + o];
            t[o] = s;
        }

        // softmax over the 8 share-channels
        float mx = t[0];
        #pragma unroll
        for (int o = 1; o < 8; ++o) mx = fmaxf(mx, t[o]);
        float esum = 0.f;
        #pragma unroll
        for (int o = 0; o < 8; ++o) { t[o] = __expf(t[o] - mx); esum += t[o]; }
        const float inv = 1.0f / esum;

        // out[o] += (v + p)[o] * w[o % 8]
        acc0 += (vv.x + p0) * t[c0 & 7] * inv;
        acc1 += (vv.y + p1) * t[c1 & 7] * inv;
    }

    float2 res;
    res.x = acc0;
    res.y = acc1;
    ((float2*)out)[(size_t)n * 32 + lane] = res;
}

// ---------------------------------------------------------------------------
extern "C" void kernel_launch(void* const* d_in, const int* in_sizes, int n_in,
                              void* d_out, int out_size, void* d_ws, size_t ws_size,
                              hipStream_t stream) {
    const float* point = (const float*)d_in[0];
    const float* feat  = (const float*)d_in[1];
    const int*   idx   = (const int*)  d_in[2];
    const float* Wq    = (const float*)d_in[3];
    const float* bq    = (const float*)d_in[4];
    const float* Wk    = (const float*)d_in[5];
    const float* bk    = (const float*)d_in[6];
    const float* Wv    = (const float*)d_in[7];
    const float* bv    = (const float*)d_in[8];
    const float* p_w1  = (const float*)d_in[9];
    const float* p_b1  = (const float*)d_in[10];
    const float* p_bn_g = (const float*)d_in[11];
    const float* p_bn_b = (const float*)d_in[12];
    const float* p_bn_m = (const float*)d_in[13];
    const float* p_bn_v = (const float*)d_in[14];
    const float* p_w2  = (const float*)d_in[15];
    const float* p_b2  = (const float*)d_in[16];
    const float* bn1_g = (const float*)d_in[17];
    const float* bn1_b = (const float*)d_in[18];
    const float* bn1_m = (const float*)d_in[19];
    const float* bn1_v = (const float*)d_in[20];
    const float* w_w1  = (const float*)d_in[21];
    const float* w_b1  = (const float*)d_in[22];
    const float* bn2_g = (const float*)d_in[23];
    const float* bn2_b = (const float*)d_in[24];
    const float* bn2_m = (const float*)d_in[25];
    const float* bn2_v = (const float*)d_in[26];
    const float* w_w2  = (const float*)d_in[27];
    const float* w_b2  = (const float*)d_in[28];

    const int n = in_sizes[0] / 3;                 // 65536 points

    // Workspace: fq | fk | fv, each n*64 fp32 (48 MB total for N=65536).
    float* fq = (float*)d_ws;
    float* fk = fq + (size_t)n * 64;
    float* fv = fk + (size_t)n * 64;

    qkv_wmma_kernel<<<n / 256, 256, 0, stream>>>(feat, Wq, bq, Wk, bk, Wv, bv,
                                                 fq, fk, fv);

    attn_kernel<<<n / 8, 256, 0, stream>>>(
        point, idx, fq, fk, fv,
        p_w1, p_b1, p_bn_g, p_bn_b, p_bn_m, p_bn_v, p_w2, p_b2,
        bn1_g, bn1_b, bn1_m, bn1_v, w_w1, w_b1,
        bn2_g, bn2_b, bn2_m, bn2_v, w_w2, w_b2,
        (float*)d_out);
}
